// GameCraftVAEAttention_58591943852383
// MI455X (gfx1250) — compile-verified, matrix operand order checked by reading, not run on hardware
//
#include <hip/hip_runtime.h>
#include <hip/hip_bf16.h>

typedef __bf16 bf16;
typedef __attribute__((ext_vector_type(8)))  __bf16 bf16x8;
typedef __attribute__((ext_vector_type(16))) __bf16 bf16x16;
typedef __attribute__((ext_vector_type(8)))  float  f32x8;

#define NB   2
#define NS   4096
#define NC   512
#define NH   8
#define ND   64
#define NG   32
#define CPG  (NC / NG)     // 16 channels per group

// ---------------------------------------------------------------------------
// WMMA fragment loader: row-major bf16 matrix, leading dim `ld` (elements).
// Per CDNA5 ISA 16-bit A/B layout: lane half h picks K = k0+8h+{0..7} and
// K = k0+16+8h+{0..7}  -> two contiguous 16-byte loads per lane.
// ---------------------------------------------------------------------------
__device__ __forceinline__ bf16x16 load_frag(const bf16* __restrict__ p,
                                             int ld, int row, int k0) {
  const int lane  = threadIdx.x & 31;
  const int half8 = (lane >> 4) * 8;
  const bf16* base = p + (size_t)row * (size_t)ld + (size_t)(k0 + half8);
  bf16x8 lo = *reinterpret_cast<const bf16x8*>(base);
  bf16x8 hi = *reinterpret_cast<const bf16x8*>(base + 16);
  return __builtin_shufflevector(lo, hi, 0,1,2,3,4,5,6,7,8,9,10,11,12,13,14,15);
}

__device__ __forceinline__ f32x8 wmma_bf16(bf16x16 a, bf16x16 b, f32x8 c) {
  return __builtin_amdgcn_wmma_f32_16x16x32_bf16(false, a, false, b,
                                                 (short)0, c, false, false);
}

// ---------------------------------------------------------------------------
// 1) GroupNorm statistics: one block per (b, g); reduce over 16ch x 4096seq.
// ---------------------------------------------------------------------------
__global__ __launch_bounds__(256)
void gn_stats_kernel(const float* __restrict__ x, float* __restrict__ stats) {
  const int bg = blockIdx.x;            // 0 .. NB*NG-1
  const int b  = bg / NG;
  const int g  = bg % NG;
  const float* base = x + (size_t)b * NS * NC + g * CPG;
  float sum = 0.f, sumsq = 0.f;
  for (int i = threadIdx.x; i < NS * CPG; i += blockDim.x) {
    int s = i / CPG, c = i % CPG;
    float v = base[(size_t)s * NC + c];
    sum += v; sumsq += v * v;
  }
  __shared__ float sh0[256], sh1[256];
  sh0[threadIdx.x] = sum; sh1[threadIdx.x] = sumsq;
  __syncthreads();
  for (int off = 128; off > 0; off >>= 1) {
    if (threadIdx.x < (unsigned)off) {
      sh0[threadIdx.x] += sh0[threadIdx.x + off];
      sh1[threadIdx.x] += sh1[threadIdx.x + off];
    }
    __syncthreads();
  }
  if (threadIdx.x == 0) {
    const float n = (float)(NS * CPG);
    float mean = sh0[0] / n;
    float var  = sh1[0] / n - mean * mean;
    stats[bg * 2 + 0] = mean;
    stats[bg * 2 + 1] = rsqrtf(var + 1e-6f);
  }
}

// ---------------------------------------------------------------------------
// 2) Apply GroupNorm affine, cast to bf16 row-major [NB*NS, NC].
// ---------------------------------------------------------------------------
__global__ __launch_bounds__(256)
void gn_apply_kernel(const float* __restrict__ x, const float* __restrict__ stats,
                     const float* __restrict__ w, const float* __restrict__ bb,
                     bf16* __restrict__ xn) {
  size_t i = (size_t)blockIdx.x * blockDim.x + threadIdx.x;   // < NB*NS*NC
  int c = (int)(i % NC);
  int b = (int)(i / ((size_t)NS * NC));
  int g = c / CPG;
  float mean = stats[(b * NG + g) * 2 + 0];
  float rstd = stats[(b * NG + g) * 2 + 1];
  float v = (x[i] - mean) * rstd * w[c] + bb[c];
  xn[i] = (bf16)v;
}

// ---------------------------------------------------------------------------
// 3) Weight transpose + cast: Wt[n][k] = W[k][n]  (512x512)
// ---------------------------------------------------------------------------
__global__ __launch_bounds__(256)
void wt_kernel(const float* __restrict__ W, bf16* __restrict__ Wt) {
  int i = blockIdx.x * blockDim.x + threadIdx.x;   // < 512*512
  int n = i / NC, k = i % NC;
  Wt[i] = (bf16)W[(size_t)k * NC + n];
}

// ---------------------------------------------------------------------------
// 4) QKV projection GEMM: [8192x512] @ [512x512] via WMMA bf16.
//    Each wave: 32(M) x 64(N) tile -> per k-step 2 A-frags x 4 B-frags
//    feed 8 independent WMMAs (better FLOPs/byte, two chains per B-frag).
//    mode 0: Q -> [B,H,S,D], pre-scaled by 1/sqrt(D) (bias included)
//    mode 1: K -> [B,H,S,D]
//    mode 2: V -> [B,H,D,S]  (transposed for the P*V B-operand)
// ---------------------------------------------------------------------------
__global__ __launch_bounds__(256)
void gemm_qkv_kernel(const bf16* __restrict__ X, const bf16* __restrict__ Wt,
                     const float* __restrict__ bias, bf16* __restrict__ out,
                     int mode, float scale) {
  const int wid  = (int)((blockIdx.x * blockDim.x + threadIdx.x) >> 5);
  const int lane = threadIdx.x & 31;
  const int tile_m = wid >> 3;          // 0..255  (32 rows each)
  const int tile_n = (wid & 7) * 64;    // 0..448
  const int row_a0 = tile_m * 32 + (lane & 15);
  f32x8 acc[2][4] = {};
  for (int k0 = 0; k0 < NC; k0 += 32) {
    // prefetch next k-step of the A rows (streams once from HBM)
    __builtin_prefetch(X + (size_t)row_a0 * NC + k0 + 32, 0, 3);
    bf16x16 a0 = load_frag(X, NC, row_a0,      k0);
    bf16x16 a1 = load_frag(X, NC, row_a0 + 16, k0);
#pragma unroll
    for (int j = 0; j < 4; ++j) {
      bf16x16 bb = load_frag(Wt, NC, tile_n + j * 16 + (lane & 15), k0);
      acc[0][j] = wmma_bf16(a0, bb, acc[0][j]);
      acc[1][j] = wmma_bf16(a1, bb, acc[1][j]);
    }
  }
  const int half = lane >> 4;
#pragma unroll
  for (int i = 0; i < 2; ++i) {
#pragma unroll
    for (int j = 0; j < 4; ++j) {
      int n = tile_n + j * 16 + (lane & 15);
      int h = n >> 6, d = n & 63;
#pragma unroll
      for (int r = 0; r < 8; ++r) {
        int m = tile_m * 32 + i * 16 + r + 8 * half;  // token index
        int b = m >> 12;                              // m / NS
        int s = m & (NS - 1);
        float v = (acc[i][j][r] + bias[n]) * scale;
        size_t idx;
        if (mode == 2) idx = ((size_t)(b * NH + h) * ND + d) * NS + s;
        else           idx = ((size_t)(b * NH + h) * NS + s) * ND + d;
        out[idx] = (bf16)v;
      }
    }
  }
}

// ---------------------------------------------------------------------------
// 5) Flash attention. grid = (B*H) * (S/16/8) = 512 blocks, 8 waves/block.
//    Each wave: 16-row Q tile; loop 32-key blocks; 4 WMMA (scores) +
//    online softmax (shfl_xor within 16-lane halves) + LDS transpose of P +
//    4 WMMA (P*V).
// ---------------------------------------------------------------------------
__global__ __launch_bounds__(256)
void attn_kernel(const bf16* __restrict__ Q, const bf16* __restrict__ K,
                 const bf16* __restrict__ Vt, bf16* __restrict__ O) {
  __shared__ __attribute__((aligned(16))) bf16 pbuf[8][16][32];
  const int wv   = threadIdx.x >> 5;
  const int lane = threadIdx.x & 31;
  const int half = lane >> 4;
  const int bh     = blockIdx.x >> 5;     // 0..15  (b*NH + h)
  const int qchunk = blockIdx.x & 31;     // 0..31
  const int q0 = (qchunk * 8 + wv) * 16;

  const bf16* Qb = Q  + (size_t)bh * NS * ND;
  const bf16* Kb = K  + (size_t)bh * NS * ND;
  const bf16* Vb = Vt + (size_t)bh * ND * NS;

  // Q fragments resident for the whole loop (D=64 -> two K=32 chunks)
  bf16x16 aq0 = load_frag(Qb, ND, q0 + (lane & 15), 0);
  bf16x16 aq1 = load_frag(Qb, ND, q0 + (lane & 15), 32);

  float mrow[8], lrow[8];
#pragma unroll
  for (int r = 0; r < 8; ++r) { mrow[r] = -1e30f; lrow[r] = 0.f; }
  f32x8 o[4] = {};

  for (int kb = 0; kb < NS; kb += 32) {
    // prefetch next key block's K rows (one line per lane)
    __builtin_prefetch(Kb + (size_t)(kb + 32 + (lane & 15)) * ND, 0, 3);
    // ---- scores: two 16-key C-tiles, keys [kb,kb+16) and [kb+16,kb+32)
    f32x8 s0 = {}, s1 = {};
    {
      bf16x16 b0 = load_frag(Kb, ND, kb + (lane & 15), 0);
      bf16x16 b1 = load_frag(Kb, ND, kb + (lane & 15), 32);
      s0 = wmma_bf16(aq0, b0, s0);
      s0 = wmma_bf16(aq1, b1, s0);
      bf16x16 b2 = load_frag(Kb, ND, kb + 16 + (lane & 15), 0);
      bf16x16 b3 = load_frag(Kb, ND, kb + 16 + (lane & 15), 32);
      s1 = wmma_bf16(aq0, b2, s1);
      s1 = wmma_bf16(aq1, b3, s1);
    }
    // ---- online softmax, per C/D row r (row M = r + 8*half, N = lane&15)
#pragma unroll
    for (int r = 0; r < 8; ++r) {
      float tmx = fmaxf(s0[r], s1[r]);
#pragma unroll
      for (int msk = 1; msk < 16; msk <<= 1)
        tmx = fmaxf(tmx, __shfl_xor(tmx, msk, 32));
      float newm = fmaxf(mrow[r], tmx);
      float p0 = __expf(s0[r] - newm);
      float p1 = __expf(s1[r] - newm);
      float rs = p0 + p1;
#pragma unroll
      for (int msk = 1; msk < 16; msk <<= 1)
        rs += __shfl_xor(rs, msk, 32);
      float corr = __expf(mrow[r] - newm);
      lrow[r] = lrow[r] * corr + rs;
      mrow[r] = newm;
#pragma unroll
      for (int j = 0; j < 4; ++j) o[j][r] *= corr;
      // stage P (C-layout -> [q][key]) into this wave's LDS tile
      pbuf[wv][r + 8 * half][(lane & 15)]      = (bf16)p0;
      pbuf[wv][r + 8 * half][16 + (lane & 15)] = (bf16)p1;
    }
    __syncthreads();
    // ---- P*V: A = P[16x32] from LDS, B = Vt rows (d), K over 32 keys
    bf16x16 ap = load_frag(&pbuf[wv][0][0], 32, lane & 15, 0);
#pragma unroll
    for (int j = 0; j < 4; ++j) {
      bf16x16 bv = load_frag(Vb, NS, j * 16 + (lane & 15), kb);
      o[j] = wmma_bf16(ap, bv, o[j]);
    }
    __syncthreads();
  }

  // epilogue: normalize by row sum (reciprocal-mul); write [B, S, H*D] bf16
  const int b = bh >> 3, h = bh & 7;
  float rinv[8];
#pragma unroll
  for (int r = 0; r < 8; ++r) rinv[r] = __frcp_rn(lrow[r]);
#pragma unroll
  for (int j = 0; j < 4; ++j) {
    int d = j * 16 + (lane & 15);
#pragma unroll
    for (int r = 0; r < 8; ++r) {
      int s = q0 + r + 8 * half;
      float v = o[j][r] * rinv[r];
      O[((size_t)b * NS + s) * (NH * ND) + h * ND + d] = (bf16)v;
    }
  }
}

// ---------------------------------------------------------------------------
// 6) Output projection + bias + residual -> fp32 d_out. 32x64 wave tiles.
// ---------------------------------------------------------------------------
__global__ __launch_bounds__(256)
void gemm_out_kernel(const bf16* __restrict__ A, const bf16* __restrict__ Wt,
                     const float* __restrict__ bias, const float* __restrict__ resid,
                     float* __restrict__ out) {
  const int wid  = (int)((blockIdx.x * blockDim.x + threadIdx.x) >> 5);
  const int lane = threadIdx.x & 31;
  const int tile_m = wid >> 3;          // 0..255
  const int tile_n = (wid & 7) * 64;
  const int row_a0 = tile_m * 32 + (lane & 15);
  f32x8 acc[2][4] = {};
  for (int k0 = 0; k0 < NC; k0 += 32) {
    __builtin_prefetch(A + (size_t)row_a0 * NC + k0 + 32, 0, 3);
    bf16x16 a0 = load_frag(A, NC, row_a0,      k0);
    bf16x16 a1 = load_frag(A, NC, row_a0 + 16, k0);
#pragma unroll
    for (int j = 0; j < 4; ++j) {
      bf16x16 bb = load_frag(Wt, NC, tile_n + j * 16 + (lane & 15), k0);
      acc[0][j] = wmma_bf16(a0, bb, acc[0][j]);
      acc[1][j] = wmma_bf16(a1, bb, acc[1][j]);
    }
  }
  const int half = lane >> 4;
#pragma unroll
  for (int i = 0; i < 2; ++i) {
#pragma unroll
    for (int j = 0; j < 4; ++j) {
      int n = tile_n + j * 16 + (lane & 15);
#pragma unroll
      for (int r = 0; r < 8; ++r) {
        int m = tile_m * 32 + i * 16 + r + 8 * half;
        size_t idx = (size_t)m * NC + n;
        out[idx] = acc[i][j][r] + bias[n] + resid[idx];
      }
    }
  }
}

// ---------------------------------------------------------------------------
extern "C" void kernel_launch(void* const* d_in, const int* in_sizes, int n_in,
                              void* d_out, int out_size, void* d_ws, size_t ws_size,
                              hipStream_t stream) {
  const float* x    = (const float*)d_in[0];
  const float* gn_w = (const float*)d_in[1];
  const float* gn_b = (const float*)d_in[2];
  const float* wq   = (const float*)d_in[3];
  const float* bq   = (const float*)d_in[4];
  const float* wk   = (const float*)d_in[5];
  const float* bk   = (const float*)d_in[6];
  const float* wvw  = (const float*)d_in[7];
  const float* bv   = (const float*)d_in[8];
  const float* wo   = (const float*)d_in[9];
  const float* bo   = (const float*)d_in[10];
  float* out = (float*)d_out;

  char* ws = (char*)d_ws;
  size_t off = 0;
  float* stats = (float*)(ws + off); off += 1024;                    // 64*2 f32
  bf16* xn     = (bf16*)(ws + off);  off += (size_t)NB*NS*NC * 2;    // 8 MB
  bf16* wqt    = (bf16*)(ws + off);  off += (size_t)NC*NC * 2;
  bf16* wkt    = (bf16*)(ws + off);  off += (size_t)NC*NC * 2;
  bf16* wvt    = (bf16*)(ws + off);  off += (size_t)NC*NC * 2;
  bf16* wot    = (bf16*)(ws + off);  off += (size_t)NC*NC * 2;
  bf16* qb     = (bf16*)(ws + off);  off += (size_t)NB*NH*NS*ND * 2; // 8 MB
  bf16* kb     = (bf16*)(ws + off);  off += (size_t)NB*NH*NS*ND * 2; // 8 MB
  bf16* vtb    = (bf16*)(ws + off);  off += (size_t)NB*NH*ND*NS * 2; // 8 MB
  bf16* attno  = xn;  // xn is dead after the QKV GEMMs -> reuse for attn out

  // 1) GroupNorm stats + apply
  gn_stats_kernel<<<NB * NG, 256, 0, stream>>>(x, stats);
  gn_apply_kernel<<<(NB * NS * NC) / 256, 256, 0, stream>>>(x, stats, gn_w, gn_b, xn);

  // 2) Weight transposes (fp32 -> bf16 [N][K])
  wt_kernel<<<(NC * NC) / 256, 256, 0, stream>>>(wq,  wqt);
  wt_kernel<<<(NC * NC) / 256, 256, 0, stream>>>(wk,  wkt);
  wt_kernel<<<(NC * NC) / 256, 256, 0, stream>>>(wvw, wvt);
  wt_kernel<<<(NC * NC) / 256, 256, 0, stream>>>(wo,  wot);

  // 3) QKV projections (WMMA). 2048 waves -> 256 blocks of 256 threads.
  gemm_qkv_kernel<<<256, 256, 0, stream>>>(xn, wqt, bq, qb,  0, 0.125f);
  gemm_qkv_kernel<<<256, 256, 0, stream>>>(xn, wkt, bk, kb,  1, 1.0f);
  gemm_qkv_kernel<<<256, 256, 0, stream>>>(xn, wvt, bv, vtb, 2, 1.0f);

  // 4) Flash attention (WMMA), writes bf16 [B,S,H*D] into attno (aliases xn)
  attn_kernel<<<512, 256, 0, stream>>>(qb, kb, vtb, attno);

  // 5) Output projection + bias + residual -> fp32 out
  gemm_out_kernel<<<256, 256, 0, stream>>>(attno, wot, bo, x, out);

  (void)in_sizes; (void)n_in; (void)out_size; (void)ws_size;
}
